// GNNtopk_75368086110726
// MI455X (gfx1250) — matrix operand order repaired and checked by reading, not compiled, not run on hardware
//
#include <hip/hip_runtime.h>
#include <hip/hip_bf16.h>

typedef __attribute__((ext_vector_type(16))) _Float16 v16h;
typedef __attribute__((ext_vector_type(8)))  float    v8f;

#define B_GRAPHS 32
#define N_NODES  2048
#define E_PER    16384
#define H_DIM    128
#define NT       (B_GRAPHS * N_NODES)   // 65536
#define E_TOT    (B_GRAPHS * E_PER)     // 524288
#define K_TOP    615                    // ceil(0.3 * 2048)

// ---------------------------------------------------------------------------
// Degree: deg[dst] += 1  (float, matches reference segment_sum of ones)
// ---------------------------------------------------------------------------
__global__ __launch_bounds__(256) void deg_kernel(const int* __restrict__ esrc,
                                                  const int* __restrict__ edst,
                                                  float* __restrict__ deg) {
    int e = blockIdx.x * 256 + threadIdx.x;
    if (e >= E_TOT) return;
    int b = e / E_PER;
    int d = b * N_NODES + edst[e];
    atomicAdd(&deg[d], 1.0f);
}

// ---------------------------------------------------------------------------
// Scatter-add messages: agg[dst] += x[src]. One thread = (edge, 4 channels).
// ---------------------------------------------------------------------------
__global__ __launch_bounds__(256) void scatter_kernel(const int* __restrict__ esrc,
                                                      const int* __restrict__ edst,
                                                      const float* __restrict__ x,
                                                      float* __restrict__ agg) {
    unsigned gid = blockIdx.x * 256u + threadIdx.x;   // E_TOT*32 threads exactly
    unsigned e   = gid >> 5;
    unsigned cg  = (gid & 31u) << 2;
    int b = (int)(e / E_PER);
    int s = b * N_NODES + esrc[e];
    int d = b * N_NODES + edst[e];
    const float4 v = *reinterpret_cast<const float4*>(x + (size_t)s * H_DIM + cg);
    float* a = agg + (size_t)d * H_DIM + cg;
    atomicAdd(a + 0, v.x);
    atomicAdd(a + 1, v.y);
    atomicAdd(a + 2, v.z);
    atomicAdd(a + 3, v.w);
}

// ---------------------------------------------------------------------------
// WMMA GEMM: out = relu( (agg/max(deg,1)) @ Wl^T + bl + xin @ Wr^T )
// One wave = one 16x16 output tile. K = H = 128.
// ---------------------------------------------------------------------------
__global__ __launch_bounds__(256) void sage_gemm_kernel(const float* __restrict__ agg,
                                                        const float* __restrict__ deg,
                                                        const float* __restrict__ xin,
                                                        const float* __restrict__ Wl,
                                                        const float* __restrict__ bl,
                                                        const float* __restrict__ Wr,
                                                        float* __restrict__ out) {
    const int K = H_DIM;
    int wave = blockIdx.x * 8 + (threadIdx.x >> 5);   // 0..32767
    int m0 = (wave >> 3) << 4;                        // 4096 M-tiles
    int n0 = (wave & 7) << 4;                         // 8 N-tiles
    int lane = threadIdx.x & 31;
    int lrow = lane & 15;
    int hi   = lane >> 4;

    int m    = m0 + lrow;                             // A row this lane loads
    int ncol = n0 + lrow;                             // B column this lane loads
    float invd = 1.0f / fmaxf(deg[m], 1.0f);

    const float* aBase  = agg + (size_t)m * K + (hi ? 8 : 0);
    const float* xBase  = xin + (size_t)m * K + (hi ? 8 : 0);
    const float* wlBase = Wl + (size_t)ncol * K + (hi ? 8 : 0);
    const float* wrBase = Wr + (size_t)ncol * K + (hi ? 8 : 0);

    v8f acc = {};
#pragma unroll
    for (int kk = 0; kk < K; kk += 32) {
        v16h a1, a2, b1, b2;
#pragma unroll
        for (int t = 0; t < 8; ++t) {
            // A 16x32 f16 layout: lanes0-15 K={0..7,16..23}, lanes16-31 K={8..15,24..31}
            a1[t]     = (_Float16)(aBase[kk + t] * invd);
            a1[8 + t] = (_Float16)(aBase[kk + 16 + t] * invd);
            a2[t]     = (_Float16)(xBase[kk + t]);
            a2[8 + t] = (_Float16)(xBase[kk + 16 + t]);
        }
#pragma unroll
        for (int t = 0; t < 16; ++t) {
            // B 32x16 f16 layout: lane = column; lanes0-15 K={0..7,16..23}, lanes16-31 K={8..15,24..31}
            // B[k][n] = W[n][k] -> two runs of 8 of row-major W
            b1[t] = (_Float16)(wlBase[kk + t + (t >= 8 ? 8 : 0)]);
            b2[t] = (_Float16)(wrBase[kk + t + (t >= 8 ? 8 : 0)]);
        }
        acc = __builtin_amdgcn_wmma_f32_16x16x32_f16(false, a1, false, b1,
                                                     (short)0, acc, false, false);
        acc = __builtin_amdgcn_wmma_f32_16x16x32_f16(false, a2, false, b2,
                                                     (short)0, acc, false, false);
    }

    float bias = bl[ncol];
#pragma unroll
    for (int r = 0; r < 8; ++r) {
        // C/D layout: VGPR r -> row m0+r (lanes0-15) / m0+r+8 (lanes16-31), col = n0+(lane&15)
        int row = m0 + r + (hi ? 8 : 0);
        float v = acc[r] + bias;
        out[(size_t)row * H_DIM + ncol] = fmaxf(v, 0.0f);
    }
}

// ---------------------------------------------------------------------------
// score[i] = tanh( dot(x[i], w) / ||w|| )
// ---------------------------------------------------------------------------
__global__ __launch_bounds__(256) void score_kernel(const float* __restrict__ x,
                                                    const float* __restrict__ pw,
                                                    float* __restrict__ sc) {
    __shared__ float w[H_DIM];
    __shared__ float red[H_DIM];
    int tid = threadIdx.x;
    if (tid < H_DIM) {
        float v = pw[tid];
        w[tid] = v;
        red[tid] = v * v;
    }
    __syncthreads();
    for (int s = 64; s > 0; s >>= 1) {
        if (tid < s) red[tid] += red[tid + s];
        __syncthreads();
    }
    float invn = 1.0f / sqrtf(red[0]);
    int node = blockIdx.x * 256 + tid;
    const float* xr = x + (size_t)node * H_DIM;
    float dot = 0.0f;
#pragma unroll 4
    for (int c = 0; c < H_DIM; ++c) dot += xr[c] * w[c];
    sc[node] = tanhf(dot * invn);
}

// ---------------------------------------------------------------------------
// Per-graph top-615: bitonic sort 2048 scores descending in LDS, then
// emb[b] = mean over kept nodes of x[node] * score.
// ---------------------------------------------------------------------------
__global__ __launch_bounds__(1024) void topk_kernel(const float* __restrict__ sc,
                                                    const float* __restrict__ x,
                                                    float* __restrict__ emb) {
    __shared__ float sv[N_NODES];
    __shared__ int   si[N_NODES];
    __shared__ float accs[8][H_DIM];
    int b = blockIdx.x;
    int tid = threadIdx.x;

    for (int i = tid; i < N_NODES; i += 1024) {
        sv[i] = sc[b * N_NODES + i];
        si[i] = i;
    }
    __syncthreads();

    // bitonic sort, descending
    for (int ksz = 2; ksz <= N_NODES; ksz <<= 1) {
        for (int j = ksz >> 1; j > 0; j >>= 1) {
            for (int i = tid; i < N_NODES; i += 1024) {
                int ixj = i ^ j;
                if (ixj > i) {
                    bool up = ((i & ksz) == 0);
                    float a = sv[i], c = sv[ixj];
                    bool sw = up ? (a < c) : (a > c);
                    if (sw) {
                        sv[i] = c; sv[ixj] = a;
                        int t = si[i]; si[i] = si[ixj]; si[ixj] = t;
                    }
                }
            }
            __syncthreads();
        }
    }

    // gated mean over top K_TOP nodes
    int part = tid >> 7;     // 0..7
    int c    = tid & 127;    // channel
    float local = 0.0f;
    for (int i = part; i < K_TOP; i += 8) {
        int node = b * N_NODES + si[i];
        local += x[(size_t)node * H_DIM + c] * sv[i];
    }
    accs[part][c] = local;
    __syncthreads();
    if (tid < H_DIM) {
        float s = 0.0f;
#pragma unroll
        for (int p = 0; p < 8; ++p) s += accs[p][tid];
        emb[b * H_DIM + tid] = s / (float)K_TOP;
    }
}

// ---------------------------------------------------------------------------
// Classifier: out[b] = relu(emb[b] @ Wc1^T + bc1) @ Wc2^T + bc2
// ---------------------------------------------------------------------------
__global__ __launch_bounds__(128) void cls_kernel(const float* __restrict__ emb,
                                                  const float* __restrict__ Wc1,
                                                  const float* __restrict__ bc1,
                                                  const float* __restrict__ Wc2,
                                                  const float* __restrict__ bc2,
                                                  float* __restrict__ out) {
    __shared__ float e[H_DIM];
    __shared__ float red[H_DIM];
    int b = blockIdx.x, c = threadIdx.x;
    e[c] = emb[b * H_DIM + c];
    __syncthreads();
    const float* wr = Wc1 + (size_t)c * H_DIM;
    float h = 0.0f;
#pragma unroll 4
    for (int k = 0; k < H_DIM; ++k) h += e[k] * wr[k];
    h = fmaxf(h + bc1[c], 0.0f);
    red[c] = h * Wc2[c];
    __syncthreads();
    for (int s = 64; s > 0; s >>= 1) {
        if (c < s) red[c] += red[c + s];
        __syncthreads();
    }
    if (c == 0) out[b] = red[0] + bc2[0];
}

// ---------------------------------------------------------------------------
extern "C" void kernel_launch(void* const* d_in, const int* in_sizes, int n_in,
                              void* d_out, int out_size, void* d_ws, size_t ws_size,
                              hipStream_t stream) {
    const float* x0  = (const float*)d_in[0];
    const int*   esr = (const int*)d_in[1];
    const int*   eds = (const int*)d_in[2];
    const float* Wl1 = (const float*)d_in[3];
    const float* bl1 = (const float*)d_in[4];
    const float* Wr1 = (const float*)d_in[5];
    const float* Wl2 = (const float*)d_in[6];
    const float* bl2 = (const float*)d_in[7];
    const float* Wr2 = (const float*)d_in[8];
    const float* pw  = (const float*)d_in[9];
    const float* Wc1 = (const float*)d_in[10];
    const float* bc1 = (const float*)d_in[11];
    const float* Wc2 = (const float*)d_in[12];
    const float* bc2 = (const float*)d_in[13];
    float* out = (float*)d_out;

    char* ws = (char*)d_ws;
    const size_t NODE_BUF = (size_t)NT * H_DIM * sizeof(float);   // 32 MB
    float* agg = (float*)ws;                 ws += NODE_BUF;
    float* x1  = (float*)ws;                 ws += NODE_BUF;
    float* x2  = (float*)ws;                 ws += NODE_BUF;
    float* deg = (float*)ws;                 ws += (size_t)NT * sizeof(float);
    float* sc  = (float*)ws;                 ws += (size_t)NT * sizeof(float);
    float* emb = (float*)ws;                 ws += (size_t)B_GRAPHS * H_DIM * sizeof(float);

    const int scatterBlocks = (E_TOT * 32) / 256;   // 65536
    const int gemmBlocks    = (NT / 16) * (H_DIM / 16) / 8;   // 4096 (8 waves/block)

    // ---- Layer 1 ----
    hipMemsetAsync(agg, 0, NODE_BUF, stream);
    hipMemsetAsync(deg, 0, (size_t)NT * sizeof(float), stream);
    deg_kernel<<<E_TOT / 256, 256, 0, stream>>>(esr, eds, deg);
    scatter_kernel<<<scatterBlocks, 256, 0, stream>>>(esr, eds, x0, agg);
    sage_gemm_kernel<<<gemmBlocks, 256, 0, stream>>>(agg, deg, x0, Wl1, bl1, Wr1, x1);

    // ---- Layer 2 (degrees identical; only re-zero agg) ----
    hipMemsetAsync(agg, 0, NODE_BUF, stream);
    scatter_kernel<<<scatterBlocks, 256, 0, stream>>>(esr, eds, x1, agg);
    sage_gemm_kernel<<<gemmBlocks, 256, 0, stream>>>(agg, deg, x1, Wl2, bl2, Wr2, x2);

    // ---- TopK pooling + classifier ----
    score_kernel<<<NT / 256, 256, 0, stream>>>(x2, pw, sc);
    topk_kernel<<<B_GRAPHS, 1024, 0, stream>>>(sc, x2, emb);
    cls_kernel<<<B_GRAPHS, 128, 0, stream>>>(emb, Wc1, bc1, Wc2, bc2, out);
}
